// NetEGNN_acce2_44822278701383
// MI455X (gfx1250) — compile-verified, matrix operand order checked by reading, not run on hardware
//
#include <hip/hip_runtime.h>
#include <hip/hip_bf16.h>

#define B_ 4
#define N_ 4096
#define K_ 32
#define H_ 64
#define L_ 3

typedef __attribute__((ext_vector_type(16))) _Float16 v16h;
typedef __attribute__((ext_vector_type(8)))  _Float16 v8h;
typedef __attribute__((ext_vector_type(4)))  _Float16 v4h;
typedef __attribute__((ext_vector_type(8)))  float    v8f;

__device__ __forceinline__ float silu_f(float x) {
    return x / (1.f + __expf(-x));
}

// Wave-private LDS phase ordering: DS ops are in-order per wave, so a single
// dscnt==0 wait makes prior ds_stores visible to subsequent ds_loads.
__device__ __forceinline__ void lds_wait() {
    asm volatile("s_wait_dscnt 0" ::: "memory");
}

// Load one 16-lane fragment (A or B) from a row-major f16 LDS tile.
// Per ISA layout for 16-bit 16x16x32 operands: lane L holds row (L&15),
// element e maps to K = (e/8)*16 + (L>>4)*8 + (e%8)  -> two b128 loads.
__device__ __forceinline__ v16h frag_ld(const _Float16* base, int stride,
                                        int row, int kbase, int half) {
    const v8h* p0 = (const v8h*)(base + row * stride + kbase + half * 8);
    const v8h* p1 = (const v8h*)(base + row * stride + kbase + 16 + half * 8);
    v8h a = *p0, b = *p1;
    v16h r;
#pragma unroll
    for (int i = 0; i < 8; ++i) { r[i] = a[i]; r[i + 8] = b[i]; }
    return r;
}

// ---------------------------------------------------------------------------
// Embedding + state init: h_st = h0 @ emb_w + emb_b ; (x,v) from ped_features
// ---------------------------------------------------------------------------
__global__ __launch_bounds__(256)
void egnn_embed_kernel(const float* __restrict__ ped, const float* __restrict__ kemb,
                       const float* __restrict__ embw, const float* __restrict__ embb,
                       float* __restrict__ h0, float* __restrict__ p0) {
    int gid = blockIdx.x * 256 + threadIdx.x;
    if (gid >= B_ * N_ * H_) return;
    int node = gid >> 6, o = gid & 63;
    const float* pf = ped + (size_t)node * 6;
    float n1 = sqrtf(pf[2] * pf[2] + pf[3] * pf[3]);
    float n2 = sqrtf(pf[4] * pf[4] + pf[5] * pf[5]);
    const float* ke = kemb + (size_t)node * 3;
    float h = embb[o] + n1 * embw[o] + n2 * embw[64 + o] +
              ke[0] * embw[128 + o] + ke[1] * embw[192 + o] + ke[2] * embw[256 + o];
    h0[(size_t)node * 64 + o] = h;
    if (o < 4) p0[(size_t)node * 4 + o] = pf[o];  // x, y, vx, vy
}

// ---------------------------------------------------------------------------
// One EGNN layer. 1 WG = 16 nodes x 32 neighbors; 8 waves, 2 nodes per wave.
// ---------------------------------------------------------------------------
#define WREG_BYTES 7680
#define SMEM_BYTES (20480 + 8192 + 8192 + 1024 + 4096 + 8 * WREG_BYTES)

__global__ __launch_bounds__(256)
void egnn_layer_kernel(
    const float* __restrict__ ped, const int* __restrict__ nmask, const int* __restrict__ nidx,
    const float* __restrict__ hin, float* __restrict__ hout,
    const float* __restrict__ pin, float* __restrict__ pout,
    const float* __restrict__ few1, const float* __restrict__ feb1,
    const float* __restrict__ few2, const float* __restrict__ feb2,
    const float* __restrict__ fxw1, const float* __restrict__ fxb1,
    const float* __restrict__ fxw2, const float* __restrict__ fxb2,
    const float* __restrict__ faw1, const float* __restrict__ fab1,
    const float* __restrict__ faw2, const float* __restrict__ fab2,
    const float* __restrict__ fhw1, const float* __restrict__ fhb1,
    const float* __restrict__ fhw2, const float* __restrict__ fhb2,
    int layer, int is_last, float* __restrict__ outp) {
    extern __shared__ char smem[];
    _Float16* W1t  = (_Float16*)smem;        // [64][160] transposed f_e_w1 (k-padded)
    _Float16* W2t  = W1t + 64 * 160;         // [64][64]  transposed f_e_w2
    _Float16* XW1t = W2t + 64 * 64;          // [64][64]  transposed f_x_w1
    float* sb1  = (float*)(XW1t + 64 * 64);  // 64
    float* sb2  = sb1 + 64;
    float* sxb1 = sb2 + 64;
    float* sxw2 = sxb1 + 64;
    float* smi  = sxw2 + 64;                 // [16][64] m_i per node
    char*  wbase = (char*)(smi + 16 * 64);   // per-wave regions

    const int tid = threadIdx.x;
    const int lane = tid & 31;
    const int wave = tid >> 5;

    // Slice per-layer weights
    few1 += (size_t)layer * (2 * H_ + 1) * H_;  feb1 += layer * H_;
    few2 += (size_t)layer * H_ * H_;            feb2 += layer * H_;
    fxw1 += (size_t)layer * H_ * H_;            fxb1 += layer * H_;
    fxw2 += (size_t)layer * H_;
    const float fxb2v = fxb2[layer];
    faw1 += (size_t)layer * H_ * H_;            fab1 += layer * H_;
    faw2 += (size_t)layer * H_;
    const float fab2v = fab2[layer];
    fhw1 += (size_t)layer * 2 * H_ * H_;        fhb1 += layer * H_;
    fhw2 += (size_t)layer * H_ * H_;            fhb2 += layer * H_;

    // Stage edge-MLP weights into LDS (transposed, f16)
    for (int i = tid; i < 64 * 160; i += 256) {
        int n = i / 160, k = i % 160;
        W1t[i] = (k < 2 * H_ + 1) ? (_Float16)few1[k * H_ + n] : (_Float16)0.f;
    }
    for (int i = tid; i < 64 * 64; i += 256) {
        int n = i >> 6, k = i & 63;
        W2t[i]  = (_Float16)few2[k * H_ + n];
        XW1t[i] = (_Float16)fxw1[k * H_ + n];
    }
    for (int i = tid; i < 64; i += 256) {
        sb1[i] = feb1[i]; sb2[i] = feb2[i]; sxb1[i] = fxb1[i]; sxw2[i] = fxw2[i];
    }
    __syncthreads();

    const int wg = blockIdx.x;
    const int b = wg / (N_ / 16);
    const int node_base = (wg % (N_ / 16)) * 16;

    _Float16* Atile = (_Float16*)(wbase + wave * WREG_BYTES);  // [16][160]
    _Float16* htile = Atile + 16 * 160;                        // [16][64]
    float* relxl = (float*)(htile + 16 * 64);                  // [16]
    float* relyl = relxl + 16;                                 // [16]
    float* gl    = relyl + 16;                                 // [64]

    const int mrow = lane & 15;
    const int hlf  = lane >> 4;

    for (int nl = 0; nl < 2; ++nl) {
        const int node_local = wave * 2 + nl;
        const int node_g = node_base + node_local;
        const size_t noff = (size_t)(b * N_) + node_g;
        const float* hself = hin + noff * H_;
        const float* pself = pin + noff * 4;

        float aggx = 0.f, aggy = 0.f;
        float miacc[4] = {0.f, 0.f, 0.f, 0.f};

        for (int kt = 0; kt < 2; ++kt) {
            // ---- stage A tile: [h_self(64) | h_neigh(64) | dist | pad] f16 ----
            {
                const int rr = lane >> 1;       // edge row 0..15
                const int hs = lane & 1;        // which 32-col half
                const int k  = kt * 16 + rr;
                const size_t eoff = noff * K_ + k;
                const int ii = nidx[eoff];
                const int mk = nmask[eoff] > 0;
                const float* hn = hin + ((size_t)(b * N_) + ii) * H_;
                _Float16* arow = Atile + rr * 160;
#pragma unroll
                for (int j = 0; j < 32; j += 4) {
                    float4 vs = *(const float4*)(hself + hs * 32 + j);
                    v4h t0 = {(_Float16)vs.x, (_Float16)vs.y, (_Float16)vs.z, (_Float16)vs.w};
                    *(v4h*)(arow + hs * 32 + j) = t0;
                    float4 vn;
                    if (mk) vn = *(const float4*)(hn + hs * 32 + j);
                    else    vn = make_float4(0.f, 0.f, 0.f, 0.f);
                    v4h t1 = {(_Float16)vn.x, (_Float16)vn.y, (_Float16)vn.z, (_Float16)vn.w};
                    *(v4h*)(arow + 64 + hs * 32 + j) = t1;
                }
                if (hs == 0) {
                    float rx = 0.f, ry = 0.f;
                    if (mk) {
                        const float* pn = pin + ((size_t)(b * N_) + ii) * 4;
                        rx = pn[0] - pself[0]; ry = pn[1] - pself[1];
                    }
                    float dist = sqrtf(rx * rx + ry * ry + 1e-12f);
                    v4h z = {(_Float16)0.f, (_Float16)0.f, (_Float16)0.f, (_Float16)0.f};
#pragma unroll
                    for (int j = 128; j < 160; j += 4) *(v4h*)(arow + j) = z;
                    arow[128] = (_Float16)dist;
                    relxl[rr] = rx; relyl[rr] = ry;
                }
            }
            lds_wait();

            // ---- GEMM1: [16x160] @ W1t -> h1, silu ----
            v8f acc[4] = {};
#pragma unroll
            for (int kc = 0; kc < 5; ++kc) {
                v16h af = frag_ld(Atile, 160, mrow, kc * 32, hlf);
#pragma unroll
                for (int t = 0; t < 4; ++t) {
                    v16h bf = frag_ld(W1t, 160, t * 16 + mrow, kc * 32, hlf);
                    acc[t] = __builtin_amdgcn_wmma_f32_16x16x32_f16(
                        false, af, false, bf, (short)0, acc[t], false, false);
                }
            }
#pragma unroll
            for (int t = 0; t < 4; ++t) {
                float bias = sb1[t * 16 + mrow];
#pragma unroll
                for (int r = 0; r < 8; ++r) acc[t][r] = silu_f(acc[t][r] + bias);
            }
#pragma unroll
            for (int t = 0; t < 4; ++t)
#pragma unroll
                for (int r = 0; r < 8; ++r)
                    htile[(r + 8 * hlf) * 64 + t * 16 + mrow] = (_Float16)acc[t][r];
            lds_wait();

            // ---- GEMM2: h1 @ W2t -> m_ij, silu ----
            v8f mm[4] = {};
#pragma unroll
            for (int kc = 0; kc < 2; ++kc) {
                v16h af = frag_ld(htile, 64, mrow, kc * 32, hlf);
#pragma unroll
                for (int t = 0; t < 4; ++t) {
                    v16h bf = frag_ld(W2t, 64, t * 16 + mrow, kc * 32, hlf);
                    mm[t] = __builtin_amdgcn_wmma_f32_16x16x32_f16(
                        false, af, false, bf, (short)0, mm[t], false, false);
                }
            }
#pragma unroll
            for (int t = 0; t < 4; ++t) {
                float bias = sb2[t * 16 + mrow];
#pragma unroll
                for (int r = 0; r < 8; ++r) mm[t][r] = silu_f(mm[t][r] + bias);
            }
            // m_i partial (column sums over 16 edge rows of this tile)
#pragma unroll
            for (int t = 0; t < 4; ++t) {
                float s = 0.f;
#pragma unroll
                for (int r = 0; r < 8; ++r) s += mm[t][r];
                s += __shfl_xor(s, 16, 32);
                miacc[t] += s;
            }
#pragma unroll
            for (int t = 0; t < 4; ++t)
#pragma unroll
                for (int r = 0; r < 8; ++r)
                    htile[(r + 8 * hlf) * 64 + t * 16 + mrow] = (_Float16)mm[t][r];
            lds_wait();

            // ---- fx hidden: m_ij @ XW1t, silu ----
            v8f fh[4] = {};
#pragma unroll
            for (int kc = 0; kc < 2; ++kc) {
                v16h af = frag_ld(htile, 64, mrow, kc * 32, hlf);
#pragma unroll
                for (int t = 0; t < 4; ++t) {
                    v16h bf = frag_ld(XW1t, 64, t * 16 + mrow, kc * 32, hlf);
                    fh[t] = __builtin_amdgcn_wmma_f32_16x16x32_f16(
                        false, af, false, bf, (short)0, fh[t], false, false);
                }
            }
#pragma unroll
            for (int t = 0; t < 4; ++t) {
                float bias = sxb1[t * 16 + mrow];
#pragma unroll
                for (int r = 0; r < 8; ++r) fh[t][r] = silu_f(fh[t][r] + bias);
            }
            // fx = hidden @ fxw2 + b2 (per edge row), reduce across the 16 N-lanes
            float p[8];
#pragma unroll
            for (int r = 0; r < 8; ++r) {
                float s = 0.f;
#pragma unroll
                for (int t = 0; t < 4; ++t) s += fh[t][r] * sxw2[t * 16 + mrow];
                p[r] = s;
            }
#pragma unroll
            for (int msk = 1; msk < 16; msk <<= 1)
#pragma unroll
                for (int r = 0; r < 8; ++r) p[r] += __shfl_xor(p[r], msk, 32);
            // agg partial: sum rel * fx over the 16 edges (rel already 0 if masked)
            float ax = 0.f, ay = 0.f;
#pragma unroll
            for (int r = 0; r < 8; ++r) {
                float fx = p[r] + fxb2v;
                int row = r + 8 * hlf;
                ax += fx * relxl[row];
                ay += fx * relyl[row];
            }
            ax += __shfl_xor(ax, 16, 32);
            ay += __shfl_xor(ay, 16, 32);
            aggx += ax; aggy += ay;
        }  // kt

        // ---- node stage (per wave, its own node) ----
        if (lane < 16) {
#pragma unroll
            for (int t = 0; t < 4; ++t) smi[node_local * 64 + t * 16 + lane] = miacc[t];
        }
        lds_wait();

        // neigh_num (K == warpSize == 32)
        float nn = (nmask[noff * K_ + lane] > 0) ? 1.f : 0.f;
        for (int msk = 1; msk < 32; msk <<= 1) nn += __shfl_xor(nn, msk, 32);
        const float inv = 1.f / (nn + 1e-6f);
        const float gx = aggx * inv, gy = aggy * inv;

        // f_a: 64->64 (silu) ->1 ; each lane handles 2 hidden dims
        float hjA = fab1[lane], hjB = fab1[lane + 32];
        for (int d = 0; d < 64; ++d) {
            float hv = hself[d];
            hjA += hv * faw1[d * 64 + lane];
            hjB += hv * faw1[d * 64 + lane + 32];
        }
        hjA = silu_f(hjA); hjB = silu_f(hjB);
        float fp = hjA * faw2[lane] + hjB * faw2[lane + 32];
        for (int msk = 1; msk < 32; msk <<= 1) fp += __shfl_xor(fp, msk, 32);
        const float fa = fp + fab2v;

        if (lane == 0) {
            const float* pd = ped + noff * 6;
            float anx  = fa * pd[4] + gx;
            float any_ = fa * pd[5] + gy;
            float vnx = pself[2] + anx, vny = pself[3] + any_;
            float xnx = pself[0] + vnx, xny = pself[1] + vny;
            float* po = pout + noff * 4;
            po[0] = xnx; po[1] = xny; po[2] = vnx; po[3] = vny;
            if (is_last) { outp[noff * 2 + 0] = anx; outp[noff * 2 + 1] = any_; }
        }

        // f_h: [h_st | m_i] (128) -> 64 (silu) -> 64, residual add
        float g1 = fhb1[lane], g2 = fhb1[lane + 32];
        for (int d = 0; d < 64; ++d) {
            float hv = hself[d];
            g1 += hv * fhw1[d * 64 + lane];
            g2 += hv * fhw1[d * 64 + lane + 32];
        }
        const float* miN = smi + node_local * 64;
        for (int d = 0; d < 64; ++d) {
            float mv = miN[d];
            g1 += mv * fhw1[(64 + d) * 64 + lane];
            g2 += mv * fhw1[(64 + d) * 64 + lane + 32];
        }
        g1 = silu_f(g1); g2 = silu_f(g2);
        gl[lane] = g1; gl[lane + 32] = g2;
        lds_wait();
        float o1 = fhb2[lane], o2 = fhb2[lane + 32];
        for (int j = 0; j < 64; ++j) {
            float gv = gl[j];
            o1 += gv * fhw2[j * 64 + lane];
            o2 += gv * fhw2[j * 64 + lane + 32];
        }
        float* ho = hout + noff * H_;
        ho[lane]      = hself[lane] + o1;
        ho[lane + 32] = hself[lane + 32] + o2;
    }  // nl
}

extern "C" void kernel_launch(void* const* d_in, const int* in_sizes, int n_in,
                              void* d_out, int out_size, void* d_ws, size_t ws_size,
                              hipStream_t stream) {
    (void)in_sizes; (void)n_in; (void)out_size; (void)ws_size;
    const float* ped  = (const float*)d_in[0];
    const int*   nmsk = (const int*)d_in[1];
    const int*   nidx = (const int*)d_in[2];
    const float* kemb = (const float*)d_in[3];
    const float* embw = (const float*)d_in[4];
    const float* embb = (const float*)d_in[5];
    const float* few1 = (const float*)d_in[6];
    const float* feb1 = (const float*)d_in[7];
    const float* few2 = (const float*)d_in[8];
    const float* feb2 = (const float*)d_in[9];
    const float* fxw1 = (const float*)d_in[10];
    const float* fxb1 = (const float*)d_in[11];
    const float* fxw2 = (const float*)d_in[12];
    const float* fxb2 = (const float*)d_in[13];
    const float* faw1 = (const float*)d_in[14];
    const float* fab1 = (const float*)d_in[15];
    const float* faw2 = (const float*)d_in[16];
    const float* fab2 = (const float*)d_in[17];
    const float* fhw1 = (const float*)d_in[18];
    const float* fhb1 = (const float*)d_in[19];
    const float* fhw2 = (const float*)d_in[20];
    const float* fhb2 = (const float*)d_in[21];

    // Workspace: h_st ping-pong (fp32) + (x,v) ping-pong
    float* hA = (float*)d_ws;
    float* hB = hA + (size_t)B_ * N_ * H_;
    float* pA = hB + (size_t)B_ * N_ * H_;
    float* pB = pA + (size_t)B_ * N_ * 4;

    const int nthr = B_ * N_ * H_;
    egnn_embed_kernel<<<(nthr + 255) / 256, 256, 0, stream>>>(ped, kemb, embw, embb, hA, pA);

    dim3 grid(B_ * (N_ / 16)), block(256);
    for (int l = 0; l < L_; ++l) {
        const float* hi = (l & 1) ? hB : hA;  float* ho = (l & 1) ? hA : hB;
        const float* pi = (l & 1) ? pB : pA;  float* po = (l & 1) ? pA : pB;
        egnn_layer_kernel<<<grid, block, SMEM_BYTES, stream>>>(
            ped, nmsk, nidx, hi, ho, pi, po,
            few1, feb1, few2, feb2, fxw1, fxb1, fxw2, fxb2,
            faw1, fab1, faw2, fab2, fhw1, fhb1, fhw2, fhb2,
            l, (l == L_ - 1) ? 1 : 0, (float*)d_out);
    }
}
